// AttentionModel_89163521064964
// MI455X (gfx1250) — matrix operand order repaired and checked by reading, not compiled, not run on hardware
//
#include <hip/hip_runtime.h>

#define B_   512
#define N_   513
#define D_   256
#define T_   128
#define H_   8
#define HD_  32
#define E_   512   // U+1
#define KSC_ 768   // D + U + 1
#define NEG_ (-1000000000.0f)

typedef __attribute__((ext_vector_type(16))) __bf16 v16bf;
typedef __attribute__((ext_vector_type(8)))  float  v8f;

// ---------------- init carry state ----------------
__global__ void init_state(float* ctx, unsigned char* mask) {
  int i = blockIdx.x * blockDim.x + threadIdx.x;
  int stride = gridDim.x * blockDim.x;
  for (int k = i; k < B_ * D_; k += stride) ctx[k] = 0.f;
  for (int k = i; k < B_ * N_; k += stride) mask[k] = 0;
}

// ---------------- one-time: fp32 -> bf16 weight conversion ----------------
__global__ void conv_bf16(const float* __restrict__ src, __bf16* __restrict__ dst, int n) {
  int i = blockIdx.x * blockDim.x + threadIdx.x;
  int stride = gridDim.x * blockDim.x;
  for (int k = i; k < n; k += stride) dst[k] = (__bf16)src[k];
}

// ---------------- graph mean + fixed context ----------------
__global__ void mean_fixed(const float* __restrict__ node,
                           const float* __restrict__ w_fc,
                           const float* __restrict__ b_fc,
                           float* __restrict__ fixed_ctx) {
  __shared__ float ge[D_];
  const int b = blockIdx.x, d = threadIdx.x;
  float s = 0.f;
  const float* p = node + (size_t)b * N_ * D_ + d;
  for (int n = 0; n < N_; ++n) s += p[(size_t)n * D_];
  ge[d] = s * (1.0f / (float)N_);
  __syncthreads();
  float f = b_fc[d];
  for (int k = 0; k < D_; ++k) f += ge[k] * w_fc[k * D_ + d];
  fixed_ctx[b * D_ + d] = f;
}

// ---- shared WMMA fragment helpers (16-bit A 16x32 layout / B lane=K-row layout) ----
__device__ __forceinline__ v16bf load_a_f32(const float* ap) {  // ap = row base + k0 + grp*8
  v16bf a;
  #pragma unroll
  for (int j = 0; j < 8; ++j) a[j]     = (__bf16)ap[j];
  #pragma unroll
  for (int j = 0; j < 8; ++j) a[8 + j] = (__bf16)ap[16 + j];
  return a;
}
__device__ __forceinline__ v16bf load_b_bf(const __bf16* bp) {  // bp = W + (k0 + grp*8)*D_ + n0 + lr
  v16bf b;
  #pragma unroll
  for (int j = 0; j < 16; ++j) b[j] = bp[(size_t)(j + (j & 8)) * D_];
  return b;
}

// ---------------- KVL projection: bf16 WMMA GEMM ----------------
// C(B*N, 768) = A(B*N,256) x W(256,768) + bias ; outputs split to K/V/L (bf16)
__global__ void kvl_gemm(const float* __restrict__ A,
                         const float* __restrict__ W,
                         const float* __restrict__ bias,
                         __bf16* __restrict__ Kb,
                         __bf16* __restrict__ Vb,
                         __bf16* __restrict__ Lb) {
  const int tid  = threadIdx.x;
  const int wave = tid >> 5, lane = tid & 31;
  const int tile = blockIdx.x * 8 + wave;
  const int tn = tile % 48, tm = tile / 48;   // 48 N-tiles over 768 cols
  const int m0 = tm * 16, n0 = tn * 16;
  const int grp = lane >> 4, lr = lane & 15;

  v8f c = {};
  const float* arow = A + (size_t)(m0 + lr) * D_;
  #pragma unroll
  for (int k0 = 0; k0 < D_; k0 += 32) {
    v16bf a, bf;
    const float* ap = arow + k0 + grp * 8;
    #pragma unroll
    for (int j = 0; j < 8; ++j) a[j]     = (__bf16)ap[j];
    #pragma unroll
    for (int j = 0; j < 8; ++j) a[8 + j] = (__bf16)ap[16 + j];
    const float* bp = W + (size_t)(k0 + grp * 8) * (3 * D_) + n0 + lr;
    #pragma unroll
    for (int j = 0; j < 16; ++j) bf[j] = (__bf16)bp[(size_t)(j + (j & 8)) * (3 * D_)];
    c = __builtin_amdgcn_wmma_f32_16x16x32_bf16(false, a, false, bf, (short)0, c,
                                                false, false);
  }
  const int col = n0 + lr;
  __bf16* dst = (col < 256) ? Kb : (col < 512 ? Vb : Lb);
  const int cc = col & 255;
  const float bv = bias[col];
  #pragma unroll
  for (int i = 0; i < 8; ++i) {
    const int row = m0 + i + 8 * grp;
    dst[(size_t)row * D_ + cc] = (__bf16)(c[i] + bv);
  }
}

// ---------------- per-step: q = fixed_ctx + [ctx, ew_t] @ w_sc + b_sc  (WMMA) ----------------
__global__ void q_gemm(const float* __restrict__ ctx,
                       const float* __restrict__ ew,
                       const __bf16* __restrict__ Wb,      // (768,256) bf16
                       const float* __restrict__ fixed_ctx,
                       const float* __restrict__ b_sc,
                       float* __restrict__ q, int t) {
  const int tid  = threadIdx.x;
  const int wave = tid >> 5, lane = tid & 31;
  const int tile = blockIdx.x * 8 + wave;     // 512 tiles: 32 M x 16 N
  const int tn = tile & 15, tm = tile >> 4;
  const int m0 = tm * 16, n0 = tn * 16;
  const int grp = lane >> 4, lr = lane & 15;
  const int row = m0 + lr;

  v8f c = {};
  #pragma unroll
  for (int k0 = 0; k0 < D_; k0 += 32) {
    v16bf a  = load_a_f32(ctx + (size_t)row * D_ + k0 + grp * 8);
    v16bf bf = load_b_bf(Wb + (size_t)(k0 + grp * 8) * D_ + n0 + lr);
    c = __builtin_amdgcn_wmma_f32_16x16x32_bf16(false, a, false, bf, (short)0, c,
                                                false, false);
  }
  const float* erow = ew + ((size_t)row * T_ + t) * E_;
  #pragma unroll
  for (int k0 = D_; k0 < KSC_; k0 += 32) {
    v16bf a  = load_a_f32(erow + (k0 - D_) + grp * 8);
    v16bf bf = load_b_bf(Wb + (size_t)(k0 + grp * 8) * D_ + n0 + lr);
    c = __builtin_amdgcn_wmma_f32_16x16x32_bf16(false, a, false, bf, (short)0, c,
                                                false, false);
  }
  const int col = n0 + lr;
  #pragma unroll
  for (int i = 0; i < 8; ++i) {
    const int r2 = m0 + i + 8 * grp;
    q[(size_t)r2 * D_ + col] = c[i] + fixed_ctx[r2 * D_ + col] + b_sc[col];
  }
}

// ---------------- per-step: attention, one WG per (b,h) ----------------
__global__ void attn_step(const float* __restrict__ q,
                          const __bf16* __restrict__ Kb, const __bf16* __restrict__ Vb,
                          const unsigned char* __restrict__ mask,
                          float* __restrict__ heads) {
  __shared__ float qh[HD_];
  __shared__ float attn[N_];
  __shared__ float red[256];
  __shared__ float Mv, Sv;
  const int b = blockIdx.x >> 3, h = blockIdx.x & 7;
  const int tid = threadIdx.x;
  const int d = tid & 31, part = tid >> 5;

  if (tid < HD_) qh[tid] = q[b * D_ + h * HD_ + tid];
  __syncthreads();
  for (int n = tid; n < N_; n += 256) {
    const __bf16* kp = Kb + ((size_t)b * N_ + n) * D_ + h * HD_;
    float s = 0.f;
    #pragma unroll
    for (int j = 0; j < HD_; ++j) s += qh[j] * (float)kp[j];
    s *= 0.17677669529663689f;  // 1/sqrt(32)
    if (mask[b * N_ + n]) s = NEG_;
    attn[n] = s;
  }
  __syncthreads();
  float m = -3.4e38f;
  for (int n = tid; n < N_; n += 256) m = fmaxf(m, attn[n]);
  red[tid] = m; __syncthreads();
  for (int s = 128; s > 0; s >>= 1) { if (tid < s) red[tid] = fmaxf(red[tid], red[tid + s]); __syncthreads(); }
  if (tid == 0) Mv = red[0];
  __syncthreads();
  float se = 0.f;
  for (int n = tid; n < N_; n += 256) { float e = __expf(attn[n] - Mv); attn[n] = e; se += e; }
  red[tid] = se; __syncthreads();
  for (int s = 128; s > 0; s >>= 1) { if (tid < s) red[tid] += red[tid + s]; __syncthreads(); }
  if (tid == 0) Sv = red[0];
  __syncthreads();
  float acc = 0.f;
  for (int n = part; n < N_; n += 8)
    acc += attn[n] * (float)Vb[((size_t)b * N_ + n) * D_ + h * HD_ + d];
  red[tid] = acc; __syncthreads();
  if (part == 0) {
    float s = 0.f;
    for (int p = 0; p < 8; ++p) s += red[p * 32 + d];
    heads[b * D_ + h * HD_ + d] = s / Sv;
  }
}

// ---------------- per-step: glimpse = heads @ w_out + b_out  (WMMA) ----------------
__global__ void glimpse_gemm(const float* __restrict__ heads,
                             const __bf16* __restrict__ Wb,   // (256,256) bf16
                             const float* __restrict__ b_out,
                             float* __restrict__ glimpse) {
  const int tid  = threadIdx.x;
  const int wave = tid >> 5, lane = tid & 31;
  const int tile = blockIdx.x * 8 + wave;     // 512 tiles: 32 M x 16 N
  const int tn = tile & 15, tm = tile >> 4;
  const int m0 = tm * 16, n0 = tn * 16;
  const int grp = lane >> 4, lr = lane & 15;
  const int row = m0 + lr;

  v8f c = {};
  #pragma unroll
  for (int k0 = 0; k0 < D_; k0 += 32) {
    v16bf a  = load_a_f32(heads + (size_t)row * D_ + k0 + grp * 8);
    v16bf bf = load_b_bf(Wb + (size_t)(k0 + grp * 8) * D_ + n0 + lr);
    c = __builtin_amdgcn_wmma_f32_16x16x32_bf16(false, a, false, bf, (short)0, c,
                                                false, false);
  }
  const int col = n0 + lr;
  #pragma unroll
  for (int i = 0; i < 8; ++i) {
    const int r2 = m0 + i + 8 * grp;
    glimpse[(size_t)r2 * D_ + col] = c[i] + b_out[col];
  }
}

// ---------------- per-step: logits, log-softmax, argmax, mask (VALU/control) ----------------
__global__ void finalize_step(const __bf16* __restrict__ Lb,
                              const float* __restrict__ glimpse,
                              unsigned char* __restrict__ mask,
                              float* __restrict__ out,
                              int* __restrict__ sel_out,
                              int t) {
  __shared__ float gs[D_], lg[N_], red[256];
  __shared__ int redi[256];
  __shared__ float Mv, logZ;
  __shared__ int selS;
  const int b = blockIdx.x, tid = threadIdx.x;

  gs[tid] = glimpse[b * D_ + tid];
  __syncthreads();
  for (int n = tid; n < N_; n += 256) {
    const __bf16* lp = Lb + ((size_t)b * N_ + n) * D_;
    float s = 0.f;
    for (int j = 0; j < D_; ++j) s += gs[j] * (float)lp[j];
    float x = 10.0f * tanhf(s * 0.0625f);   // CLIP * tanh(logit / sqrt(D))
    if (mask[b * N_ + n]) x = NEG_;
    lg[n] = x;
  }
  __syncthreads();
  float m = -3.4e38f;
  for (int n = tid; n < N_; n += 256) m = fmaxf(m, lg[n]);
  red[tid] = m; __syncthreads();
  for (int s = 128; s > 0; s >>= 1) { if (tid < s) red[tid] = fmaxf(red[tid], red[tid + s]); __syncthreads(); }
  if (tid == 0) Mv = red[0];
  __syncthreads();
  float se = 0.f;
  for (int n = tid; n < N_; n += 256) se += __expf(lg[n] - Mv);
  red[tid] = se; __syncthreads();
  for (int s = 128; s > 0; s >>= 1) { if (tid < s) red[tid] += red[tid + s]; __syncthreads(); }
  if (tid == 0) logZ = Mv + logf(red[0]);
  __syncthreads();
  for (int n = tid; n < N_; n += 256)
    out[((size_t)b * T_ + t) * N_ + n] = lg[n] - logZ;
  // argmax, first occurrence on ties
  float bv = -3.4e38f; int bi = 0;
  for (int n = tid; n < N_; n += 256) if (lg[n] > bv) { bv = lg[n]; bi = n; }
  red[tid] = bv; redi[tid] = bi;
  __syncthreads();
  for (int s = 128; s > 0; s >>= 1) {
    if (tid < s) {
      float v2 = red[tid + s]; int i2 = redi[tid + s];
      if (v2 > red[tid] || (v2 == red[tid] && i2 < redi[tid])) { red[tid] = v2; redi[tid] = i2; }
    }
    __syncthreads();
  }
  if (tid == 0) { selS = redi[0]; sel_out[b] = redi[0]; }
  __syncthreads();
  const int sel = selS;
  for (int n = tid; n < N_; n += 256)
    if (n == sel && sel > 0) mask[b * N_ + n] = 1;
}

// ---------------- per-step: edge_emb GEMM + ctx EMA update  (WMMA, gathered A) ----------------
// edge = [node[b,sel[b]] || node[b,N-1]](512x512) @ w_edge(512x256) + b_edge
// ctx  = ctx + (edge - ctx)/t  (each element owned by exactly one lane)
__global__ void edge_gemm(const float* __restrict__ node,
                          const int* __restrict__ sel,
                          const __bf16* __restrict__ Wb,   // (512,256) bf16
                          const float* __restrict__ b_edge,
                          float* __restrict__ ctx, float inv_t) {
  const int tid  = threadIdx.x;
  const int wave = tid >> 5, lane = tid & 31;
  const int tile = blockIdx.x * 8 + wave;     // 512 tiles: 32 M x 16 N
  const int tn = tile & 15, tm = tile >> 4;
  const int m0 = tm * 16, n0 = tn * 16;
  const int grp = lane >> 4, lr = lane & 15;
  const int row = m0 + lr;                    // batch index b for this lane's A rows

  const int sb = sel[row];
  const float* srow = node + ((size_t)row * N_ + sb) * D_;        // selected embedding
  const float* crow = node + ((size_t)row * N_ + (N_ - 1)) * D_;  // current (last node)

  v8f c = {};
  #pragma unroll
  for (int k0 = 0; k0 < D_; k0 += 32) {       // K in [0,256): sel_emb
    v16bf a  = load_a_f32(srow + k0 + grp * 8);
    v16bf bf = load_b_bf(Wb + (size_t)(k0 + grp * 8) * D_ + n0 + lr);
    c = __builtin_amdgcn_wmma_f32_16x16x32_bf16(false, a, false, bf, (short)0, c,
                                                false, false);
  }
  #pragma unroll
  for (int k0 = D_; k0 < 2 * D_; k0 += 32) {  // K in [256,512): curr_emb
    v16bf a  = load_a_f32(crow + (k0 - D_) + grp * 8);
    v16bf bf = load_b_bf(Wb + (size_t)(k0 + grp * 8) * D_ + n0 + lr);
    c = __builtin_amdgcn_wmma_f32_16x16x32_bf16(false, a, false, bf, (short)0, c,
                                                false, false);
  }
  const int col = n0 + lr;
  #pragma unroll
  for (int i = 0; i < 8; ++i) {
    const int r2 = m0 + i + 8 * grp;
    const float e = c[i] + b_edge[col];
    const float co = ctx[(size_t)r2 * D_ + col];
    ctx[(size_t)r2 * D_ + col] = co + (e - co) * inv_t;
  }
}

extern "C" void kernel_launch(void* const* d_in, const int* in_sizes, int n_in,
                              void* d_out, int out_size, void* d_ws, size_t ws_size,
                              hipStream_t stream) {
  const float* node   = (const float*)d_in[0];
  const float* ew     = (const float*)d_in[1];
  const float* w_kvl  = (const float*)d_in[2];
  const float* b_kvl  = (const float*)d_in[3];
  const float* w_fc   = (const float*)d_in[4];
  const float* b_fc   = (const float*)d_in[5];
  const float* w_sc   = (const float*)d_in[6];
  const float* b_sc   = (const float*)d_in[7];
  const float* w_out  = (const float*)d_in[8];
  const float* b_out  = (const float*)d_in[9];
  const float* w_edge = (const float*)d_in[10];
  const float* b_edge = (const float*)d_in[11];
  float* out = (float*)d_out;

  char* w = (char*)d_ws;
  size_t off = 0;
  const size_t kvlN = (size_t)B_ * N_ * D_;
  __bf16* Kb = (__bf16*)(w + off); off += kvlN * 2;
  __bf16* Vb = (__bf16*)(w + off); off += kvlN * 2;
  __bf16* Lb = (__bf16*)(w + off); off += kvlN * 2;
  float* fixed_ctx = (float*)(w + off); off += (size_t)B_ * D_ * 4;
  float* ctx       = (float*)(w + off); off += (size_t)B_ * D_ * 4;
  float* heads     = (float*)(w + off); off += (size_t)B_ * D_ * 4;
  float* qbuf      = (float*)(w + off); off += (size_t)B_ * D_ * 4;
  float* glimpse   = (float*)(w + off); off += (size_t)B_ * D_ * 4;
  __bf16* wsc_bf   = (__bf16*)(w + off); off += (size_t)KSC_ * D_ * 2;
  __bf16* wout_bf  = (__bf16*)(w + off); off += (size_t)D_ * D_ * 2;
  __bf16* wedge_bf = (__bf16*)(w + off); off += (size_t)2 * D_ * D_ * 2;
  int* selbuf      = (int*)(w + off);    off += (size_t)B_ * 4;
  unsigned char* mask = (unsigned char*)(w + off); off += (size_t)B_ * N_;
  (void)ws_size; (void)in_sizes; (void)n_in; (void)out_size;

  init_state<<<1024, 256, 0, stream>>>(ctx, mask);
  conv_bf16<<<256, 256, 0, stream>>>(w_sc, wsc_bf, KSC_ * D_);
  conv_bf16<<<256, 256, 0, stream>>>(w_out, wout_bf, D_ * D_);
  conv_bf16<<<256, 256, 0, stream>>>(w_edge, wedge_bf, 2 * D_ * D_);
  mean_fixed<<<B_, 256, 0, stream>>>(node, w_fc, b_fc, fixed_ctx);
  kvl_gemm<<<(16416 * 48) / 8, 256, 0, stream>>>(node, w_kvl, b_kvl, Kb, Vb, Lb);
  for (int t = 0; t < T_; ++t) {
    const float inv_t = 1.0f / (float)(t + 1);
    q_gemm<<<64, 256, 0, stream>>>(ctx, ew, wsc_bf, fixed_ctx, b_sc, qbuf, t);
    attn_step<<<B_ * H_, 256, 0, stream>>>(qbuf, Kb, Vb, mask, heads);
    glimpse_gemm<<<64, 256, 0, stream>>>(heads, wout_bf, b_out, glimpse);
    finalize_step<<<B_, 256, 0, stream>>>(Lb, glimpse, mask, out, selbuf, t);
    edge_gemm<<<64, 256, 0, stream>>>(node, selbuf, wedge_bf, b_edge, ctx, inv_t);
  }
}
